// MultiWaySelfAttention_79688823210164
// MI455X (gfx1250) — matrix operand order repaired and checked by reading, not compiled, not run on hardware
//
#include <hip/hip_runtime.h>
#include <hip/hip_bf16.h>

// MI455X / gfx1250 wave32 WMMA implementation of the MultiWay transformer block.
// All GEMMs + attention on v_wmma_f32_16x16x32_bf16. Global->LDS staging uses
// CDNA5 async-to-LDS ops (ASYNCcnt) with double-buffered LDS (1 barrier/iter).

typedef __bf16 bf16;
typedef bf16 v16bf __attribute__((ext_vector_type(16)));
typedef float v8f __attribute__((ext_vector_type(8)));
typedef __attribute__((address_space(3))) bf16 lds_bf16;

#define TOK 8192   // B*S
#define DIM 768
#define SEQ 2048
#define NBATCH 4
#define NH 12
#define DHEAD 64
#define DFF 3072
#define NWAY 4

__device__ __forceinline__ bf16 f2bf(float f) {
  union { float f; unsigned u; } v; v.f = f;
  unsigned r = v.u + 0x7fffu + ((v.u >> 16) & 1u);
  unsigned short h = (unsigned short)(r >> 16);
  return __builtin_bit_cast(bf16, h);
}

__device__ __forceinline__ v8f wmma_bf(v16bf a, v16bf b, v8f c) {
  return __builtin_amdgcn_wmma_f32_16x16x32_bf16(false, a, false, b, (short)0, c,
                                                 false, false);
}

// LDS byte offset of a generic pointer to __shared__ data.
__device__ __forceinline__ unsigned lds_addr(const bf16* p) {
  return (unsigned)(unsigned long long)(const lds_bf16*)p;
}

// CDNA5 async copy: 16 bytes per lane, global -> LDS, tracked by ASYNCcnt.
__device__ __forceinline__ void async_b128(unsigned ldsoff, const bf16* g) {
  asm volatile("global_load_async_to_lds_b128 %0, %1, off"
               :: "v"(ldsoff), "v"(g) : "memory");
}
__device__ __forceinline__ void wait_async0() {
  asm volatile("s_wait_asynccnt 0x0" ::: "memory");
}

// A fragment (16x32, M x K) from row-major [M x K] source.
// Element e -> K=16*kh+e; m=lane&15, kh=lane>>4.
__device__ __forceinline__ v16bf load_a_frag(const bf16* base, int ld, int k0) {
  int lane = threadIdx.x & 31;
  const bf16* row = base + (lane & 15) * ld + k0 + ((lane >> 4) & 1) * 16;
  union { v16bf v; uint4 q[2]; } u;
  u.q[0] = *(const uint4*)row;
  u.q[1] = *(const uint4*)(row + 8);
  return u.v;
}

// B fragment (32x16, K x N) where the source stores columns contiguously,
// i.e. row-major [N x K] (= B^T). Element e -> K = k0 + 16*kh + e, n = lane&15.
__device__ __forceinline__ v16bf load_bt_frag(const bf16* base, int ld, int k0) {
  int lane = threadIdx.x & 31;
  const bf16* p = base + (lane & 15) * ld + k0 + ((lane >> 4) & 1) * 16;
  union { v16bf v; uint4 q[2]; } u;
  u.q[0] = *(const uint4*)p;
  u.q[1] = *(const uint4*)(p + 8);
  return u.v;
}

// ---------------------------------------------------------------- elementwise
// fp32 [K x N] (per expert, blockIdx.y) -> bf16 transposed [N x K]
__global__ void cvtT_kernel(const float* __restrict__ src, bf16* __restrict__ dst,
                            int K, int N) {
  int e = blockIdx.y;
  int i = blockIdx.x * blockDim.x + threadIdx.x;
  if (i < K * N) {
    int k = i / N, n = i % N;
    dst[(size_t)e * K * N + (size_t)n * K + k] =
        f2bf(src[(size_t)e * K * N + i]);
  }
}

// ---------------------------------------------------------------- layernorm
__global__ __launch_bounds__(256) void ln_kernel(const float* __restrict__ x,
                                                 const float* __restrict__ g,
                                                 const float* __restrict__ b,
                                                 bf16* __restrict__ xn) {
  const int wave = threadIdx.x >> 5, lane = threadIdx.x & 31;
  const int t = blockIdx.x * 8 + wave;
  const float* row = x + (size_t)t * DIM;
  float vals[24];
  float s = 0.f;
#pragma unroll
  for (int i = 0; i < 24; ++i) { vals[i] = row[lane + 32 * i]; s += vals[i]; }
#pragma unroll
  for (int mk = 1; mk < 32; mk <<= 1) s += __shfl_xor(s, mk, 32);
  float mu = s * (1.0f / DIM);
  float vs = 0.f;
#pragma unroll
  for (int i = 0; i < 24; ++i) { float d = vals[i] - mu; vs += d * d; }
#pragma unroll
  for (int mk = 1; mk < 32; mk <<= 1) vs += __shfl_xor(vs, mk, 32);
  float inv = rsqrtf(vs * (1.0f / DIM) + 1e-6f);
  bf16* orow = xn + (size_t)t * DIM;
#pragma unroll
  for (int i = 0; i < 24; ++i) {
    int c = lane + 32 * i;
    orow[c] = f2bf((vals[i] - mu) * inv * g[c] + b[c]);
  }
}

// ---------------------------------------------------------------- WMMA GEMM
// C[M,N] = A[M,K] @ W[K,N] + bias, W stored TRANSPOSED [N,K].
// Block tile 128x64, 8 waves each compute 16x64. Double-buffered async staging.
// modes: 1=Q/K proj (bias, bf16 out -> [B,H,S,DH])
//        2=O proj (bias + residual, f32 + bf16 out row-major)
//        3=FFN1 (expert z, gathered A rows, bias+ReLU, bf16 out compact)
//        4=FFN2 (expert z, compact A rows, bias, f32 out scattered by token)
//        5=V proj (bias, bf16 out -> [B,H,DH,S])
#define BM 128
#define BN 64
#define BK 32
#define LDA 40
#define LDWT 40

__global__ __launch_bounds__(256) void gemm_kernel(
    const bf16* __restrict__ A, const bf16* __restrict__ WT,
    const float* __restrict__ bias, int M, int N, int K, int mode,
    float* __restrict__ outF, bf16* __restrict__ outB,
    const float* __restrict__ resid, const int* __restrict__ gidx,
    const int* __restrict__ goffs, const int* __restrict__ gcnt) {
  __shared__ bf16 sA[2][BM * LDA];
  __shared__ bf16 sWT[2][BN * LDWT];
  const int nt = blockIdx.x, mt = blockIdx.y, ex = blockIdx.z;
  const int tid = threadIdx.x;
  int rowCount = M, rowBase = 0;
  const bf16* Wp = WT;
  const float* bp = bias;
  if (mode >= 3 && mode <= 4) {
    rowCount = gcnt[ex];
    if (mt * BM >= rowCount) return;
    Wp = WT + (size_t)ex * K * N;
    bp = bias + (size_t)ex * N;
    rowBase = goffs[ex];
  }
  // A staging: row = tid>>1 (0..127), 16-element segment (tid&1)*16
  const int ar = tid >> 1, as_ = (tid & 1) * 16;
  const int lr = mt * BM + ar;
  const bf16* arow;
  if (mode == 3) {
    int cr = lr < rowCount ? lr : rowCount - 1;
    arow = A + (size_t)gidx[rowBase + cr] * K;
  } else if (mode == 4) {
    int cr = lr < rowCount ? lr : rowCount - 1;
    arow = A + (size_t)(rowBase + cr) * K;
  } else {
    arow = A + (size_t)lr * K;
  }
  // W staging: row = tid>>2 (0..63), 8-element segment (tid&3)*8
  const int wr2 = tid >> 2, ws_ = (tid & 3) * 8;
  const bf16* wrow = Wp + (size_t)(nt * BN + wr2) * K;
  unsigned aoff[2] = { lds_addr(&sA[0][ar * LDA + as_]),
                       lds_addr(&sA[1][ar * LDA + as_]) };
  unsigned woff[2] = { lds_addr(&sWT[0][wr2 * LDWT + ws_]),
                       lds_addr(&sWT[1][wr2 * LDWT + ws_]) };
  auto stage = [&](int buf, int kt) {
    const bf16* a0 = arow + kt * BK + as_;
    async_b128(aoff[buf], a0);
    async_b128(aoff[buf] + 16, a0 + 8);
    async_b128(woff[buf], wrow + kt * BK + ws_);
  };
  const int wave = tid >> 5, lane = tid & 31;
  v8f c[4] = {};
  const int nK = K / BK;
  stage(0, 0);
  for (int kt = 0; kt < nK; ++kt) {
    wait_async0();          // this wave's tile-kt data landed in LDS
    __syncthreads();        // everyone's landed; everyone done reading buf^1
    if (kt + 1 < nK) stage((kt + 1) & 1, kt + 1);  // overlap with compute
    const bf16* cA = sA[kt & 1];
    const bf16* cW = sWT[kt & 1];
    v16bf af = load_a_frag(&cA[wave * 16 * LDA], LDA, 0);
#pragma unroll
    for (int j = 0; j < 4; ++j) {
      v16bf bj = load_bt_frag(&cW[(j * 16) * LDWT], LDWT, 0);
      c[j] = wmma_bf(af, bj, c[j]);
    }
  }
  const int half = lane >> 4, nloc = lane & 15;
#pragma unroll
  for (int j = 0; j < 4; ++j) {
    const int ncol = nt * BN + j * 16 + nloc;
#pragma unroll
    for (int r = 0; r < 8; ++r) {
      const int mrow = mt * BM + wave * 16 + r + 8 * half;
      float val = c[j][r];
      if (mode == 1) {
        val += bp[ncol];
        int bb = mrow >> 11, ss = mrow & (SEQ - 1);
        int hh = ncol >> 6, dd = ncol & (DHEAD - 1);
        outB[(((size_t)bb * NH + hh) * SEQ + ss) * DHEAD + dd] = f2bf(val);
      } else if (mode == 5) {
        val += bp[ncol];
        int bb = mrow >> 11, ss = mrow & (SEQ - 1);
        int hh = ncol >> 6, dd = ncol & (DHEAD - 1);
        outB[(((size_t)bb * NH + hh) * DHEAD + dd) * SEQ + ss] = f2bf(val);
      } else if (mode == 2) {
        val += bp[ncol] + resid[(size_t)mrow * DIM + ncol];
        outF[(size_t)mrow * DIM + ncol] = val;
        outB[(size_t)mrow * DIM + ncol] = f2bf(val);
      } else if (mode == 3) {
        if (mrow < rowCount) {
          val += bp[ncol];
          val = val > 0.f ? val : 0.f;
          outB[(size_t)(rowBase + mrow) * DFF + ncol] = f2bf(val);
        }
      } else {  // mode 4
        if (mrow < rowCount) {
          val += bp[ncol];
          outF[(size_t)gidx[rowBase + mrow] * DIM + ncol] = val;
        }
      }
    }
  }
}

// ---------------------------------------------------------------- attention
// q,k: [B,H,S,DH] bf16; vT: [B,H,DH,S] bf16; ctx out: [B,S,H*DH] bf16.
#define LK 72
#define LVT 40
__global__ __launch_bounds__(256) void attn_kernel(const bf16* __restrict__ q,
                                                   const bf16* __restrict__ k,
                                                   const bf16* __restrict__ vT,
                                                   bf16* __restrict__ ctx) {
  __shared__ bf16 sK[2][32 * LK];    // 32 keys x 64 dh (row-major)
  __shared__ bf16 sVT[2][64 * LVT];  // 64 dh x 32 keys (V^T tile)
  __shared__ bf16 sP[8 * 16 * 40];
  const int tid = threadIdx.x;
  const int bh = blockIdx.x >> 4;   // b*NH + h
  const int qt = blockIdx.x & 15;   // 128-row query tile
  const int wave = tid >> 5, lane = tid & 31;
  const int q0 = qt * 128 + wave * 16;
  const bf16* qb = q + ((size_t)bh * SEQ + q0) * DHEAD;
  v16bf qa0 = load_a_frag(qb, DHEAD, 0);
  v16bf qa1 = load_a_frag(qb, DHEAD, 32);
  float rm[8], rl[8];
  v8f acc0 = {}, acc1 = {}, acc2 = {}, acc3 = {};
#pragma unroll
  for (int r = 0; r < 8; ++r) { rm[r] = -3.0e38f; rl[r] = 0.f; }
  bf16* sPw = &sP[wave * 16 * 40];
  const int ki = tid >> 3, kc = (tid & 7) * 8;  // K tile copy: 32 x 64
  const int vi = tid >> 2, vc = (tid & 3) * 8;  // VT tile copy: 64 x 32
  const size_t kvbase = (size_t)bh * SEQ * DHEAD;
  const bf16* krow = k + kvbase + (size_t)ki * DHEAD + kc;
  const bf16* vrow = vT + kvbase + (size_t)vi * SEQ + vc;
  unsigned koff[2] = { lds_addr(&sK[0][ki * LK + kc]),
                       lds_addr(&sK[1][ki * LK + kc]) };
  unsigned voff[2] = { lds_addr(&sVT[0][vi * LVT + vc]),
                       lds_addr(&sVT[1][vi * LVT + vc]) };
  auto stage = [&](int buf, int kt) {
    async_b128(koff[buf], krow + (size_t)kt * 32 * DHEAD);
    async_b128(voff[buf], vrow + kt * 32);
  };
  stage(0, 0);
  for (int kt = 0; kt < SEQ / 32; ++kt) {
    wait_async0();
    __syncthreads();
    if (kt + 1 < SEQ / 32) stage((kt + 1) & 1, kt + 1);
    const bf16* cK = sK[kt & 1];
    const bf16* cV = sVT[kt & 1];
    v8f s0 = {}, s1 = {};
    s0 = wmma_bf(qa0, load_bt_frag(&cK[0], LK, 0), s0);
    s0 = wmma_bf(qa1, load_bt_frag(&cK[0], LK, 32), s0);
    s1 = wmma_bf(qa0, load_bt_frag(&cK[16 * LK], LK, 0), s1);
    s1 = wmma_bf(qa1, load_bt_frag(&cK[16 * LK], LK, 32), s1);
    const float sc = 0.125f;  // 1/sqrt(64)
    float alpha[8];
#pragma unroll
    for (int r = 0; r < 8; ++r) {
      float a0 = s0[r] * sc, a1 = s1[r] * sc;
      float t = fmaxf(a0, a1);
#pragma unroll
      for (int mk = 1; mk < 16; mk <<= 1) t = fmaxf(t, __shfl_xor(t, mk, 32));
      float nm = fmaxf(rm[r], t);
      float al = __expf(rm[r] - nm);
      rm[r] = nm;
      float p0 = __expf(a0 - nm), p1 = __expf(a1 - nm);
      float rs = p0 + p1;
#pragma unroll
      for (int mk = 1; mk < 16; mk <<= 1) rs += __shfl_xor(rs, mk, 32);
      rl[r] = rl[r] * al + rs;
      alpha[r] = al;
      int row = r + 8 * (lane >> 4);
      sPw[row * 40 + (lane & 15)] = f2bf(p0);
      sPw[row * 40 + 16 + (lane & 15)] = f2bf(p1);
    }
#pragma unroll
    for (int r = 0; r < 8; ++r) {
      acc0[r] *= alpha[r]; acc1[r] *= alpha[r];
      acc2[r] *= alpha[r]; acc3[r] *= alpha[r];
    }
    asm volatile("s_wait_dscnt 0x0" ::: "memory");  // P staging RAW within wave
    v16bf pa = load_a_frag(sPw, 40, 0);
    acc0 = wmma_bf(pa, load_bt_frag(&cV[0 * LVT], LVT, 0), acc0);
    acc1 = wmma_bf(pa, load_bt_frag(&cV[16 * LVT], LVT, 0), acc1);
    acc2 = wmma_bf(pa, load_bt_frag(&cV[32 * LVT], LVT, 0), acc2);
    acc3 = wmma_bf(pa, load_bt_frag(&cV[48 * LVT], LVT, 0), acc3);
  }
  const int bb = bh / NH, hh = bh % NH;
  const int half = lane >> 4, nloc = lane & 15;
#pragma unroll
  for (int r = 0; r < 8; ++r) {
    int srow = q0 + r + 8 * half;
    float inv = 1.0f / rl[r];
    size_t base = ((size_t)bb * SEQ + srow) * DIM + hh * DHEAD + nloc;
    ctx[base + 0]  = f2bf(acc0[r] * inv);
    ctx[base + 16] = f2bf(acc1[r] * inv);
    ctx[base + 32] = f2bf(acc2[r] * inv);
    ctx[base + 48] = f2bf(acc3[r] * inv);
  }
}

// ---------------------------------------------------------------- MoE routing
__global__ void route_init_kernel(int* counts) {
  if (threadIdx.x < NWAY) counts[threadIdx.x] = 0;
}
__global__ void route_rank_kernel(const int* __restrict__ w, int* counts,
                                  int* __restrict__ rank) {
  int t = blockIdx.x * blockDim.x + threadIdx.x;
  if (t < TOK) rank[t] = atomicAdd(&counts[w[t] & (NWAY - 1)], 1);
}
__global__ void route_offs_kernel(const int* __restrict__ counts,
                                  int* __restrict__ offs) {
  if (threadIdx.x == 0) {
    int o = 0;
    for (int e = 0; e < NWAY; ++e) { offs[e] = o; o += counts[e]; }
  }
}
__global__ void route_scatter_kernel(const int* __restrict__ w,
                                     const int* __restrict__ rank,
                                     const int* __restrict__ offs,
                                     int* __restrict__ gidx) {
  int t = blockIdx.x * blockDim.x + threadIdx.x;
  if (t < TOK) gidx[offs[w[t] & (NWAY - 1)] + rank[t]] = t;
}

// ---------------------------------------------------------------- classifier
__global__ void cls_kernel(const float* __restrict__ outF,
                           const float* __restrict__ wcls,
                           const float* __restrict__ bcls,
                           float* __restrict__ dst) {
  int i = blockIdx.x * blockDim.x + threadIdx.x;
  if (i >= TOK * NWAY) return;
  int t = i >> 2, c = i & 3;
  const float* row = outF + (size_t)t * DIM;
  float s = bcls[c];
  for (int kk = 0; kk < DIM; ++kk) s = fmaf(row[kk], wcls[kk * NWAY + c], s);
  dst[i] = s;
}

// ---------------------------------------------------------------- launch
extern "C" void kernel_launch(void* const* d_in, const int* in_sizes, int n_in,
                              void* d_out, int out_size, void* d_ws,
                              size_t ws_size, hipStream_t stream) {
  const float* x     = (const float*)d_in[0];
  const int*   wrt   = (const int*)d_in[1];
  const float* ln_g  = (const float*)d_in[2];
  const float* ln_b  = (const float*)d_in[3];
  const float* wq    = (const float*)d_in[4];
  const float* bq    = (const float*)d_in[5];
  const float* wk    = (const float*)d_in[6];
  const float* bk    = (const float*)d_in[7];
  const float* wv    = (const float*)d_in[8];
  const float* bv    = (const float*)d_in[9];
  const float* wo    = (const float*)d_in[10];
  const float* bo    = (const float*)d_in[11];
  const float* wcls  = (const float*)d_in[12];
  const float* bcls  = (const float*)d_in[13];
  const float* w1    = (const float*)d_in[14];
  const float* b1    = (const float*)d_in[15];
  const float* w2    = (const float*)d_in[16];
  const float* b2    = (const float*)d_in[17];
  (void)in_sizes; (void)n_in; (void)out_size; (void)ws_size;

  char* ws = (char*)d_ws;
  const size_t actB = (size_t)TOK * DIM * 2;  // one bf16 activation plane
  size_t o = 0;
  bf16* xnB  = (bf16*)(ws + o); o += actB;
  bf16* qB   = (bf16*)(ws + o); o += actB;
  bf16* kB   = (bf16*)(ws + o); o += actB;
  bf16* vB   = (bf16*)(ws + o); o += actB;   // stored [B,H,DH,S]
  bf16* ctxB = (bf16*)(ws + o); o += actB;
  // FFN hidden reuses the region of xn/q/k/v/ctx (all dead by then):
  bf16* hB = (bf16*)ws;  // TOK*DFF*2 = 50.3MB <= 5*actB = 62.9MB
  bf16* wqB = (bf16*)(ws + o); o += (size_t)DIM * DIM * 2;
  bf16* wkB = (bf16*)(ws + o); o += (size_t)DIM * DIM * 2;
  bf16* wvB = (bf16*)(ws + o); o += (size_t)DIM * DIM * 2;
  bf16* woB = (bf16*)(ws + o); o += (size_t)DIM * DIM * 2;
  bf16* w1B = (bf16*)(ws + o); o += (size_t)NWAY * DIM * DFF * 2;
  bf16* w2B = (bf16*)(ws + o); o += (size_t)NWAY * DFF * DIM * 2;
  float* outF = (float*)(ws + o); o += (size_t)TOK * DIM * 4;
  bf16* outB = (bf16*)(ws + o); o += actB;
  int* counts = (int*)(ws + o); o += 256;
  int* offs   = (int*)(ws + o); o += 256;
  int* rank   = (int*)(ws + o); o += (size_t)TOK * 4;
  int* gidx   = (int*)(ws + o); o += (size_t)TOK * 4;

  // 1) weights -> bf16, transposed to [N,K] per expert
  int nw = DIM * DIM;
  dim3 gw((nw + 255) / 256, 1);
  cvtT_kernel<<<gw, 256, 0, stream>>>(wq, wqB, DIM, DIM);
  cvtT_kernel<<<gw, 256, 0, stream>>>(wk, wkB, DIM, DIM);
  cvtT_kernel<<<gw, 256, 0, stream>>>(wv, wvB, DIM, DIM);
  cvtT_kernel<<<gw, 256, 0, stream>>>(wo, woB, DIM, DIM);
  int nf = DIM * DFF;
  dim3 gf((nf + 255) / 256, NWAY);
  cvtT_kernel<<<gf, 256, 0, stream>>>(w1, w1B, DIM, DFF);
  cvtT_kernel<<<gf, 256, 0, stream>>>(w2, w2B, DFF, DIM);

  // 2) layernorm
  ln_kernel<<<TOK / 8, 256, 0, stream>>>(x, ln_g, ln_b, xnB);

  // 3) QKV projections
  dim3 gp(DIM / BN, TOK / BM, 1);
  gemm_kernel<<<gp, 256, 0, stream>>>(xnB, wqB, bq, TOK, DIM, DIM, 1, nullptr,
                                      qB, nullptr, nullptr, nullptr, nullptr);
  gemm_kernel<<<gp, 256, 0, stream>>>(xnB, wkB, bk, TOK, DIM, DIM, 1, nullptr,
                                      kB, nullptr, nullptr, nullptr, nullptr);
  gemm_kernel<<<gp, 256, 0, stream>>>(xnB, wvB, bv, TOK, DIM, DIM, 5, nullptr,
                                      vB, nullptr, nullptr, nullptr, nullptr);

  // 4) flash attention
  attn_kernel<<<NBATCH * NH * (SEQ / 128), 256, 0, stream>>>(qB, kB, vB, ctxB);

  // 5) output projection + residual
  gemm_kernel<<<gp, 256, 0, stream>>>(ctxB, woB, bo, TOK, DIM, DIM, 2, outF,
                                      outB, x, nullptr, nullptr, nullptr);

  // 6) MoE routing (gather tokens per expert)
  route_init_kernel<<<1, 64, 0, stream>>>(counts);
  route_rank_kernel<<<TOK / 256, 256, 0, stream>>>(wrt, counts, rank);
  route_offs_kernel<<<1, 32, 0, stream>>>(counts, offs);
  route_scatter_kernel<<<TOK / 256, 256, 0, stream>>>(wrt, rank, offs, gidx);

  // 7) FFN expert GEMMs (grouped, early-exit on empty tiles)
  dim3 g1(DFF / BN, TOK / BM, NWAY);
  gemm_kernel<<<g1, 256, 0, stream>>>(outB, w1B, b1, TOK, DFF, DIM, 3, nullptr,
                                      hB, nullptr, gidx, offs, counts);
  dim3 g2(DIM / BN, TOK / BM, NWAY);
  gemm_kernel<<<g2, 256, 0, stream>>>(hB, w2B, b2, TOK, DIM, DFF, 4,
                                      (float*)d_out, nullptr, nullptr, gidx,
                                      offs, counts);

  // 8) way classifier head -> second output
  cls_kernel<<<(TOK * NWAY + 255) / 256, 256, 0, stream>>>(
      outF, wcls, bcls, (float*)d_out + (size_t)TOK * DIM);
}
